// NTXentLoss_82497731821883
// MI455X (gfx1250) — matrix operand order repaired and checked by reading, not compiled, not run on hardware
//
#include <hip/hip_runtime.h>
#include <hip/hip_bf16.h>
#include <hip/hip_fp16.h>

// NT-Xent loss, fused: normalize -> f16 WMMA GEMM (Z Z^T) with exp+rowsum epilogue -> log/mean.
// B=8192, D=128, T=0.5 (1/T = 2). Diagonal handled analytically (self-sim == 1 -> subtract e^2).

#define BB   8192
#define DD   128
#define TWOB 16384
#define LDS_STRIDE 136      // 128 + 8 halves pad -> 272B row stride, conflict-free ds access
#define NSPLIT 8            // N-range splits (grid.y)
#define CHUNKS_PER (128 / NSPLIT)

typedef __attribute__((ext_vector_type(16))) _Float16 v16h;
typedef __attribute__((ext_vector_type(8)))  _Float16 h8;
typedef __attribute__((ext_vector_type(8)))  float    v8f;
typedef __attribute__((ext_vector_type(4)))  float    f4;
typedef __attribute__((ext_vector_type(4)))  _Float16 h4;
typedef __attribute__((ext_vector_type(4)))  int      v4i;

// AS-qualified b128 payload types for the async DMA builtin
typedef __attribute__((address_space(1))) v4i g_v4i;   // global
typedef __attribute__((address_space(3))) v4i l_v4i;   // LDS

union HV { v16h v; h8 h[2]; };

// ---- CDNA5 async global->LDS DMA (ASYNCcnt-tracked), with safe fallback -----
#if defined(__has_builtin)
#  if __has_builtin(__builtin_amdgcn_global_load_async_to_lds_b128)
#    define HAVE_ASYNC_LDS 1
#  endif
#endif
#ifndef HAVE_ASYNC_LDS
#  define HAVE_ASYNC_LDS 0
#endif

#if HAVE_ASYNC_LDS
#  if defined(__has_builtin) && __has_builtin(__builtin_amdgcn_s_wait_asynccnt)
#    define WAIT_ASYNC() __builtin_amdgcn_s_wait_asynccnt(0)
#  else
#    define WAIT_ASYNC() asm volatile("s_wait_asynccnt 0" ::: "memory")
#  endif
#endif

// ---------------------------------------------------------------------------
// Kernel 1: L2-normalize rows of z_i, z_j -> f16 Z = [zi_n ; zj_n]; positive
// dots; zero the row-sum accumulator. One wave32 per row, 4 floats per lane.
// ---------------------------------------------------------------------------
__global__ __launch_bounds__(256) void k_norm(const float* __restrict__ zi,
                                              const float* __restrict__ zj,
                                              _Float16* __restrict__ zh,
                                              float* __restrict__ posdot,
                                              float* __restrict__ rowsum) {
    const int tid = blockIdx.x * 256 + threadIdx.x;
    if (tid < TWOB) rowsum[tid] = 0.0f;

    const int lane = threadIdx.x & 31;
    const int wave = threadIdx.x >> 5;
    const int row  = blockIdx.x * 8 + wave;

    const f4 a = ((const f4*)(zi + (size_t)row * DD))[lane];
    const f4 b = ((const f4*)(zj + (size_t)row * DD))[lane];

    float sii = a[0]*a[0] + a[1]*a[1] + a[2]*a[2] + a[3]*a[3];
    float sjj = b[0]*b[0] + b[1]*b[1] + b[2]*b[2] + b[3]*b[3];
    float sij = a[0]*b[0] + a[1]*b[1] + a[2]*b[2] + a[3]*b[3];

    #pragma unroll
    for (int off = 16; off > 0; off >>= 1) {
        sii += __shfl_xor(sii, off, 32);
        sjj += __shfl_xor(sjj, off, 32);
        sij += __shfl_xor(sij, off, 32);
    }

    const float ri = 1.0f / fmaxf(sqrtf(sii), 1e-12f);
    const float rj = 1.0f / fmaxf(sqrtf(sjj), 1e-12f);

    h4 oi, oj;
    #pragma unroll
    for (int k = 0; k < 4; ++k) {
        oi[k] = (_Float16)(a[k] * ri);
        oj[k] = (_Float16)(b[k] * rj);
    }
    *(h4*)(zh + (size_t)row        * DD + lane * 4) = oi;
    *(h4*)(zh + (size_t)(BB + row) * DD + lane * 4) = oj;

    if (lane == 0) posdot[row] = sij * ri * rj;   // cosine sim of positive pair
}

// ---------------------------------------------------------------------------
// Kernel 2: fused S = Z Z^T, epilogue exp(2*s) + row-sum.
// Grid (64, 8): x = 256-row M block (8 waves x 2 tiles of 16 rows),
//               y = 1/8 of the N range (16 chunks of 128 columns).
// Double-buffered LDS B panel; async DMA copy overlapped with WMMA compute.
// Each ds_load_b128 of B feeds 2 WMMAs (2 M-tiles share B fragments).
// ---------------------------------------------------------------------------
__global__ __launch_bounds__(256) void k_gemm(const _Float16* __restrict__ zh,
                                              float* __restrict__ rowsum) {
    __shared__ _Float16 lds[2][128 * LDS_STRIDE];   // 2 x 34.8 KB

    const int lane = threadIdx.x & 31;
    const int wave = threadIdx.x >> 5;
    const int half = lane >> 4;        // 0: lanes 0-15, 1: lanes 16-31
    const int l15  = lane & 15;
    const int Mb0  = blockIdx.x * 256 + wave * 16;
    const int Mb1  = Mb0 + 128;

    // A fragments for both M tiles: 2 x 16 rows x K=128 (documented A layout:
    // elems 0..7 = K kbase+half*8.., elems 8..15 = K kbase+16+half*8..)
    v16h afrag[2][4];
    #pragma unroll
    for (int m = 0; m < 2; ++m) {
        const int arow = (m ? Mb1 : Mb0) + l15;
        #pragma unroll
        for (int kc = 0; kc < 4; ++kc) {
            HV u;
            u.h[0] = *(const h8*)(zh + (size_t)arow * DD + kc * 32 + half * 8);
            u.h[1] = *(const h8*)(zh + (size_t)arow * DD + kc * 32 + 16 + half * 8);
            afrag[m][kc] = u.v;
        }
    }

    float acc0[8], acc1[8];
    #pragma unroll
    for (int r = 0; r < 8; ++r) { acc0[r] = 0.0f; acc1[r] = 0.0f; }

    // staging assignment: 2 threads per column, 64 halves (8x16B) each
    const int scol  = threadIdx.x >> 1;
    const int spart = (threadIdx.x & 1) * 64;
    const int ch0   = blockIdx.y * CHUNKS_PER;

    // Prologue: synchronous stage of first chunk into buffer 0.
    {
        const f4* src = (const f4*)(zh + (size_t)(ch0 * 128 + scol) * DD + spart);
        f4*       dst = (f4*)(&lds[0][scol * LDS_STRIDE + spart]);
        #pragma unroll
        for (int q = 0; q < 8; ++q) dst[q] = src[q];
    }
    __syncthreads();

    for (int i = 0; i < CHUNKS_PER; ++i) {
        const int cur = i & 1;
        const int nxt = cur ^ 1;
        const bool have_next = (i + 1) < CHUNKS_PER;
        const size_t gnext = (size_t)((ch0 + i + 1) * 128 + scol) * DD + spart;

#if HAVE_ASYNC_LDS
        if (have_next) {   // async DMA into the other buffer, overlapped with compute
            const char* gs = (const char*)(zh + gnext);
            char*       ls = (char*)(&lds[nxt][scol * LDS_STRIDE + spart]);
            #pragma unroll
            for (int q = 0; q < 8; ++q)
                __builtin_amdgcn_global_load_async_to_lds_b128(
                    (g_v4i*)(gs + q * 16),
                    (l_v4i*)(ls + q * 16),
                    0, 0);
        }
#else
        f4 pre[8];
        if (have_next) {   // register prefetch, stored to LDS after compute
            const f4* src = (const f4*)(zh + gnext);
            #pragma unroll
            for (int q = 0; q < 8; ++q) pre[q] = src[q];
        }
#endif

        // ---- compute 8 N-tiles from lds[cur]; each B fragment feeds 2 WMMAs
        const _Float16* base = &lds[cur][0];
        for (int t = 0; t < 8; ++t) {
            const _Float16* bp = base + (t * 16 + l15) * LDS_STRIDE + half * 8;
            v8f c0 = {}, c1 = {};
            #pragma unroll
            for (int kc = 0; kc < 4; ++kc) {
                HV u;   // B elems 0..7 = K kbase + half*8 + 0..7, elems 8..15 = K kbase + 16 + half*8 + 0..7
                u.h[0] = *(const h8*)(bp + kc * 32);
                u.h[1] = *(const h8*)(bp + kc * 32 + 16);
                c0 = __builtin_amdgcn_wmma_f32_16x16x32_f16(
                        false, afrag[0][kc], false, u.v, (short)0, c0, false, false);
                c1 = __builtin_amdgcn_wmma_f32_16x16x32_f16(
                        false, afrag[1][kc], false, u.v, (short)0, c1, false, false);
            }
            #pragma unroll
            for (int r = 0; r < 8; ++r) {
                acc0[r] += __expf(c0[r] * 2.0f);   // exp(s / T), T = 0.5
                acc1[r] += __expf(c1[r] * 2.0f);
            }
        }

#if HAVE_ASYNC_LDS
        WAIT_ASYNC();
        __syncthreads();
#else
        if (have_next) {
            f4* dst = (f4*)(&lds[nxt][scol * LDS_STRIDE + spart]);
            #pragma unroll
            for (int q = 0; q < 8; ++q) dst[q] = pre[q];
        }
        __syncthreads();
#endif
    }

    // Reduce across the 16 lanes of each half-wave (columns), emit row sums.
    #pragma unroll
    for (int r = 0; r < 8; ++r) {
        float v0 = acc0[r], v1 = acc1[r];
        #pragma unroll
        for (int off = 8; off > 0; off >>= 1) {
            v0 += __shfl_xor(v0, off, 16);
            v1 += __shfl_xor(v1, off, 16);
        }
        if (l15 == 0) {
            atomicAdd(&rowsum[Mb0 + r + half * 8], v0);
            atomicAdd(&rowsum[Mb1 + r + half * 8], v1);
        }
    }
}

// ---------------------------------------------------------------------------
// Kernel 3: loss = mean_b( log((rs_i - e^2) + (rs_j - e^2)) - 2*dot_b )
// ---------------------------------------------------------------------------
__global__ __launch_bounds__(256) void k_loss(const float* __restrict__ posdot,
                                              const float* __restrict__ rowsum,
                                              float* __restrict__ out) {
    __shared__ float red[8];
    const float e2 = 7.38905609893065f;   // exp(1/T), self-similarity term
    float s = 0.0f;
    for (int b = threadIdx.x; b < BB; b += 256) {
        const float denom = (rowsum[b] - e2) + (rowsum[BB + b] - e2);
        s += __logf(denom) - 2.0f * posdot[b];
    }
    #pragma unroll
    for (int off = 16; off > 0; off >>= 1) s += __shfl_xor(s, off, 32);
    if ((threadIdx.x & 31) == 0) red[threadIdx.x >> 5] = s;
    __syncthreads();
    if (threadIdx.x == 0) {
        float t = 0.0f;
        #pragma unroll
        for (int i = 0; i < 8; ++i) t += red[i];
        out[0] = t / (float)BB;
    }
}

// ---------------------------------------------------------------------------
extern "C" void kernel_launch(void* const* d_in, const int* in_sizes, int n_in,
                              void* d_out, int out_size, void* d_ws, size_t ws_size,
                              hipStream_t stream) {
    const float* zi = (const float*)d_in[0];
    const float* zj = (const float*)d_in[1];

    _Float16* zh     = (_Float16*)d_ws;                                   // 4 MB
    float*    posdot = (float*)((char*)d_ws + (size_t)TWOB * DD * 2);     // 32 KB
    float*    rowsum = posdot + BB;                                       // 64 KB
    float*    out    = (float*)d_out;

    k_norm<<<BB / 8, 256, 0, stream>>>(zi, zj, zh, posdot, rowsum);
    k_gemm<<<dim3(TWOB / 256, NSPLIT), 256, 0, stream>>>(zh, rowsum);
    k_loss<<<1, 256, 0, stream>>>(posdot, rowsum, out);
}
